// MultiHeadFALayer_13048110645522
// MI455X (gfx1250) — matrix operand order repaired and checked by reading, not compiled, not run on hardware
//
#include <hip/hip_runtime.h>
#include <hip/hip_bf16.h>

// MultiHeadFALayer for MI455X (gfx1250, wave32).
// ws layout (floats): [0, N) deg->nd ; [aggOff, aggOff+N*256) agg accumulator ;
// [aggOff+N*256, +8192) W_cat as f16 (64x256 halfs, 32KB).
// h (12.8MB) + agg (51.2MB) are L2-resident (192MB L2), so the per-edge
// gathers + 256 f32 atomic adds/edge resolve at L2 bandwidth, never HBM.

typedef __attribute__((ext_vector_type(16))) _Float16 v16h;
typedef __attribute__((ext_vector_type(8)))  float    v8f;

__global__ void zero_ws_kernel(float4* __restrict__ ws4, int n4) {
  int i = blockIdx.x * blockDim.x + threadIdx.x;
  int stride = gridDim.x * blockDim.x;
  float4 z = {0.f, 0.f, 0.f, 0.f};
  for (; i < n4; i += stride) ws4[i] = z;
}

__global__ void deg_kernel(const int* __restrict__ row, float* __restrict__ deg, int E) {
  int e = blockIdx.x * blockDim.x + threadIdx.x;
  if (e < E) atomicAdd(&deg[row[e]], 1.0f);
}

__global__ void nd_kernel(float* __restrict__ deg_nd, int n) {
  int i = blockIdx.x * blockDim.x + threadIdx.x;
  if (i < n) {
    float d = deg_nd[i];
    deg_nd[i] = rsqrtf(d < 1.f ? 1.f : d);
  }
}

// One-time f32 -> f16 conversion of W_cat (64x256), so GEMM waves load B
// directly as halfs instead of re-converting per wave.
__global__ void wc_f16_kernel(const float* __restrict__ Wc, _Float16* __restrict__ wcH, int n) {
  int i = blockIdx.x * blockDim.x + threadIdx.x;
  if (i < n) wcH[i] = (_Float16)Wc[i];
}

// Persistent wave32-per-edge kernel, grid-stride over edges.
// Gate weights/bias are wave-invariant per lane -> hoisted into registers once.
// Per edge: 2 float2 gathers, 4 butterfly reductions, tanh, 8 atomics/lane.
__global__ void edge_kernel(const float* __restrict__ h,
                            const int* __restrict__ row, const int* __restrict__ col,
                            const float* __restrict__ gw, const float* __restrict__ gb,
                            const float* __restrict__ nd, float* __restrict__ agg, int E) {
  int lane   = threadIdx.x & 31;
  int wid    = (blockIdx.x * blockDim.x + threadIdx.x) >> 5;
  int nwaves = (gridDim.x * blockDim.x) >> 5;

  float w1a[4], w1b[4], w2a[4], w2b[4], gbv[4];
#pragma unroll
  for (int k = 0; k < 4; ++k) {
    const float* g1 = gw + k * 128;          // w1 row k
    w1a[k] = g1[2 * lane];      w1b[k] = g1[2 * lane + 1];
    w2a[k] = g1[64 + 2 * lane]; w2b[k] = g1[64 + 2 * lane + 1];
    gbv[k] = gb[k];
  }

  for (int e = wid; e < E; e += nwaves) {
    int r = row[e];
    int c = col[e];
    float2 hr = *(const float2*)(h + (size_t)r * 64 + 2 * lane);
    float2 hc = *(const float2*)(h + (size_t)c * 64 + 2 * lane);
    float p[4];
#pragma unroll
    for (int k = 0; k < 4; ++k)
      p[k] = hr.x * w1a[k] + hr.y * w1b[k] + hc.x * w2a[k] + hc.y * w2b[k];
#pragma unroll
    for (int k = 0; k < 4; ++k) {
#pragma unroll
      for (int m = 16; m >= 1; m >>= 1) p[k] += __shfl_xor(p[k], m, 32);
    }
    float scale = nd[r] * nd[c];
    float* dst = agg + (size_t)c * 256 + 2 * lane;
#pragma unroll
    for (int k = 0; k < 4; ++k) {
      float nk = tanhf(p[k] + gbv[k]) * scale;
      atomicAdd(dst + k * 64,     nk * hr.x);
      atomicAdd(dst + k * 64 + 1, nk * hr.y);
    }
  }
}

// out = relu(agg(N x 256) @ Wc^T(256 x 64) + bc).
// One wave computes a 32x64 strip: two 16x32 A tiles per K-step share each
// pre-converted f16 B fragment -> 8 WMMAs per K-step, agg read once,
// Wc(f16) traffic halved, zero B-side cvt in the loop.
// Layouts used here (contiguous K order, same for A and B; not the ISA table order):
//   A 16x32 f16: lane-half hf owns K = k0+16*hf+{0..15} contiguous,
//                M = lane&15.
//   B 32x16 f16: N = lane&15, K = k0+16*hf+{0..15} contiguous.
//   C/D 16x16 f32: vgpr r -> M = r + 8*hf, N = lane&15.
__global__ void gemm_relu_wmma(const float* __restrict__ agg, const _Float16* __restrict__ wcH,
                               const float* __restrict__ bc, float* __restrict__ out, int n) {
  int wave = (blockIdx.x * blockDim.x + threadIdx.x) >> 5;
  int lane = threadIdx.x & 31;
  int row0 = wave * 32;
  if (row0 >= n) return;                    // wave-uniform: EXEC all-1s for WMMA
  int half = lane >> 4;
  int lp   = lane & 15;
  int m0 = row0 + lp;      if (m0 >= n) m0 = n - 1;   // clamps (N%16==0 anyway)
  int m1 = row0 + 16 + lp; if (m1 >= n) m1 = n - 1;
  const float* arow0 = agg + (size_t)m0 * 256;
  const float* arow1 = agg + (size_t)m1 * 256;

  v8f acc[2][4] = {};
  for (int k0 = 0; k0 < 256; k0 += 32) {
    v16h a0, a1;
    int ab = k0 + 16 * half;
#pragma unroll
    for (int i = 0; i < 8; ++i) {
      a0[i]     = (_Float16)arow0[ab + i];
      a0[i + 8] = (_Float16)arow0[ab + 8 + i];
      a1[i]     = (_Float16)arow1[ab + i];
      a1[i + 8] = (_Float16)arow1[ab + 8 + i];
    }
#pragma unroll
    for (int ct = 0; ct < 4; ++ct) {
      v16h b = *(const v16h*)(wcH + (size_t)(ct * 16 + lp) * 256 + k0 + 16 * half);
      acc[0][ct] = __builtin_amdgcn_wmma_f32_16x16x32_f16(false, a0, false, b,
                                                          (short)0, acc[0][ct], false, false);
      acc[1][ct] = __builtin_amdgcn_wmma_f32_16x16x32_f16(false, a1, false, b,
                                                          (short)0, acc[1][ct], false, false);
    }
  }

#pragma unroll
  for (int t = 0; t < 2; ++t) {
    int rt = row0 + 16 * t;
    if (rt >= n) break;                     // wave-uniform
    if (rt + 16 <= n) {                     // full tile: unguarded epilogue
#pragma unroll
      for (int ct = 0; ct < 4; ++ct) {
        float bias = bc[ct * 16 + lp];
#pragma unroll
        for (int rr = 0; rr < 8; ++rr) {
          int m = rt + rr + 8 * half;
          float v = acc[t][ct][rr] + bias;
          out[(size_t)m * 64 + ct * 16 + lp] = v > 0.f ? v : 0.f;
        }
      }
    } else {                                // cold fallback for ragged N
#pragma unroll
      for (int ct = 0; ct < 4; ++ct) {
        float bias = bc[ct * 16 + lp];
#pragma unroll
        for (int rr = 0; rr < 8; ++rr) {
          int m = rt + rr + 8 * half;
          if (m < n) {
            float v = acc[t][ct][rr] + bias;
            out[(size_t)m * 64 + ct * 16 + lp] = v > 0.f ? v : 0.f;
          }
        }
      }
    }
  }
}

extern "C" void kernel_launch(void* const* d_in, const int* in_sizes, int n_in,
                              void* d_out, int out_size, void* d_ws, size_t ws_size,
                              hipStream_t stream) {
  const float* h  = (const float*)d_in[0];
  const int*   ei = (const int*)d_in[1];
  const float* gw = (const float*)d_in[2];
  const float* gb = (const float*)d_in[3];
  const float* Wc = (const float*)d_in[4];
  const float* bc = (const float*)d_in[5];
  float* out = (float*)d_out;

  const int N = in_sizes[0] / 64;   // 50000
  const int E = in_sizes[1] / 2;    // 800000
  const int WCN = in_sizes[4];      // 64*256 = 16384

  float* ws = (float*)d_ws;
  const int aggOff = ((N + 127) / 128) * 128;
  float* deg_nd = ws;
  float* agg    = ws + aggOff;
  _Float16* wcH = (_Float16*)(ws + aggOff + (size_t)N * 256);
  const int* row = ei;
  const int* col = ei + E;

  // 1) zero deg + agg (wcH is fully overwritten by wc_f16_kernel)
  int totalFloats = aggOff + N * 256;
  zero_ws_kernel<<<2048, 256, 0, stream>>>((float4*)ws, totalFloats / 4);

  // 2) W_cat -> f16 copy (64KB read, once)
  wc_f16_kernel<<<(WCN + 255) / 256, 256, 0, stream>>>(Wc, wcH, WCN);

  // 3) degree histogram
  deg_kernel<<<(E + 255) / 256, 256, 0, stream>>>(row, deg_nd, E);

  // 4) nd = rsqrt(max(deg,1)) in place
  nd_kernel<<<(N + 255) / 256, 256, 0, stream>>>(deg_nd, N);

  // 5) per-edge gate + scatter-add: persistent waves, 8 waves/block
  edge_kernel<<<4096, 256, 0, stream>>>(h, row, col, gw, gb, deg_nd, agg, E);

  // 6) WMMA GEMM + bias + relu: one wave per 32-row strip
  int waves  = (N + 31) / 32;
  int blocks = (waves * 32 + 255) / 256;
  gemm_relu_wmma<<<blocks, 256, 0, stream>>>(agg, wcH, bc, out, N);
}
